// MultiHeadAttention_27917287424434
// MI455X (gfx1250) — matrix operand order repaired and checked by reading, not compiled, not run on hardware
//
#include <hip/hip_runtime.h>

// ---------------------------------------------------------------------------
// MI455X (gfx1250) fused bilinear multi-head attention.
//   B=2, S=512, E=1024, H=16, D=64
// Pipeline:
//   1) gemm_proj (mode 0..2): Qh/Kh bf16 [b,h,s,d], Vh^T bf16 [b,h,d,s]
//   2) qp_kernel: qp[b,h,s,f] = sum_e qh[b,h,s,e] * p[b,s,h,e,f]   (f32 VALU,
//      memory-bound on 268MB p tensor; prefetch-driven streaming)
//   3) attn: flash-style causal softmax(qp@Kh^T/8) @ Vh using v_wmma bf16,
//      with double-buffered GLOBAL_LOAD_ASYNC_TO_LDS_B128 K/V staging
//   4) gemm_proj (mode 3): out = att @ Wo^T  (f32 output)
// ---------------------------------------------------------------------------

typedef __attribute__((ext_vector_type(16))) __bf16 v16bf;
typedef __attribute__((ext_vector_type(8)))  float  v8f;

#define SEQ   512
#define EMB   1024
#define HEADS 16
#define DH    64

__device__ __forceinline__ __bf16 f2b(float f) { return (__bf16)f; }
__device__ __forceinline__ float  b2f(__bf16 b) { return (float)b; }

__device__ __forceinline__ v8f vzero8() {
  v8f z = {0.f, 0.f, 0.f, 0.f, 0.f, 0.f, 0.f, 0.f};
  return z;
}

// Async global->LDS copy of 16 bytes per lane (gfx1250, tracked by ASYNCcnt).
// Generic pointers into LDS carry the wave-relative LDS offset in their low
// 32 bits (aperture mapping).
__device__ __forceinline__ void async_copy16(void* lds_dst, const void* gsrc) {
  unsigned int ldsa = (unsigned int)(unsigned long long)lds_dst;
  asm volatile("global_load_async_to_lds_b128 %0, %1, off"
               :
               : "v"(ldsa), "v"(gsrc)
               : "memory");
}
__device__ __forceinline__ void wait_async() {
  asm volatile("s_wait_asynccnt 0" ::: "memory");
}

// A-fragment (16x32 bf16, M x K), ISA 7.12.2 layout:
//   lanes 0-15 : row M=lane, VGPR0-3 hold K=0..7,  VGPR4-7 hold K=16..23
//   lanes 16-31: row M=lane-16, VGPR0-3 hold K=8..15, VGPR4-7 hold K=24..31
__device__ __forceinline__ v16bf load_a_frag(const __bf16* base, int ld) {
  const int lane = threadIdx.x & 31;
  const int m    = lane & 15;
  const int kh   = (lane >> 4) & 1;      // which K-half this lane holds
  const __bf16* r = base + m * ld;
  v16bf f;
#pragma unroll
  for (int i = 0; i < 8; ++i) f[i] = r[kh * 8 + i];          // K 0..7 / 8..15
#pragma unroll
  for (int i = 0; i < 8; ++i) f[8 + i] = r[16 + kh * 8 + i]; // K 16..23 / 24..31
  return f;
}

// B-fragment (32x16 bf16, K x N) loaded from a K-major ("transposed") source:
// src[n][k] contiguous in k with stride ld. Lanes 0-15 hold K=0..7,16..23 of column
// n=lane, lanes 16-31 hold K=8..15,24..31 of column n=lane-16 (matches A layout).
__device__ __forceinline__ v16bf load_b_frag_t(const __bf16* base, int ld) {
  const int lane = threadIdx.x & 31;
  const int n    = lane & 15;
  const int koff = (lane >> 4) << 3;     // 0 or 8
  const __bf16* r = base + n * ld + koff;
  v16bf f;
#pragma unroll
  for (int i = 0; i < 8; ++i) f[i] = r[i];
#pragma unroll
  for (int i = 0; i < 8; ++i) f[8 + i] = r[16 + i];
  return f;
}

__device__ __forceinline__ v8f wmma_bf16(v16bf a, v16bf b, v8f c) {
  return __builtin_amdgcn_wmma_f32_16x16x32_bf16(
      /*neg_a=*/false, a, /*neg_b=*/false, b,
      /*c_mod=*/(short)0, c, /*reuse_a=*/false, /*reuse_b=*/false);
}

// ---------------------------------------------------------------------------
// GEMM: C[i,j] = sum_e A[i,e] * W[j,e]   (M=N=K=1024)
// mode 0/1: Q/K projection -> bf16 [b,h,s,d]
// mode 2  : V projection   -> bf16 [b,h,d,s] (transposed for attention B-frags)
// mode 3  : output proj (A = bf16 att matrix) -> f32 [b,s,e]
// Workgroup: 256 threads = 8 waves; tile 128(M) x 64(N), K staged 32 at a time.
// ---------------------------------------------------------------------------
__global__ __launch_bounds__(256) void gemm_proj(
    const float* __restrict__ Aq, const float* __restrict__ Ak,
    const float* __restrict__ Av, const __bf16* __restrict__ Aatt,
    const float* __restrict__ Wq, const float* __restrict__ Wk,
    const float* __restrict__ Wv, const float* __restrict__ Wo,
    __bf16* __restrict__ Qh, __bf16* __restrict__ Kh,
    __bf16* __restrict__ VhT, float* __restrict__ Out, int mode_base) {
  const int mode = mode_base + blockIdx.z;
  const float* Af = (mode == 0) ? Aq : (mode == 1) ? Ak : Av;
  const float* W  = (mode == 0) ? Wq : (mode == 1) ? Wk : (mode == 2) ? Wv : Wo;

  const int m0 = blockIdx.x * 128;
  const int n0 = blockIdx.y * 64;
  const int tid  = threadIdx.x;
  const int wv   = tid >> 5;
  const int lane = tid & 31;

  __shared__ __bf16 As[128][32];
  __shared__ __bf16 Bs[64][32];

  v8f acc[4];
#pragma unroll
  for (int n = 0; n < 4; ++n) acc[n] = vzero8();

  for (int k0 = 0; k0 < EMB; k0 += 32) {
    __syncthreads();
    // ---- stage A tile (128x32), convert f32->bf16 on the fly ----
    {
      const int row = tid >> 1;
      const int kq  = (tid & 1) * 16;
      if (mode == 3) {
        const __bf16* src = Aatt + (size_t)(m0 + row) * EMB + k0 + kq;
#pragma unroll
        for (int i = 0; i < 16; ++i) As[row][kq + i] = src[i];
      } else {
        const float* src = Af + (size_t)(m0 + row) * EMB + k0 + kq;
#pragma unroll
        for (int i = 0; i < 16; i += 4) {
          float4 v = *(const float4*)(src + i);
          As[row][kq + i + 0] = f2b(v.x);
          As[row][kq + i + 1] = f2b(v.y);
          As[row][kq + i + 2] = f2b(v.z);
          As[row][kq + i + 3] = f2b(v.w);
        }
      }
    }
    // ---- stage W tile (64 rows of W = 64 columns of W^T), K-major ----
    {
      const int row = tid >> 2;
      const int kq  = (tid & 3) * 8;
      const float* src = W + (size_t)(n0 + row) * EMB + k0 + kq;
#pragma unroll
      for (int i = 0; i < 8; i += 4) {
        float4 v = *(const float4*)(src + i);
        Bs[row][kq + i + 0] = f2b(v.x);
        Bs[row][kq + i + 1] = f2b(v.y);
        Bs[row][kq + i + 2] = f2b(v.z);
        Bs[row][kq + i + 3] = f2b(v.w);
      }
    }
    __syncthreads();

    // Preload all fragments, then issue WMMAs back-to-back.
    const v16bf a = load_a_frag(&As[wv * 16][0], 32);
    v16bf bfr[4];
#pragma unroll
    for (int n = 0; n < 4; ++n) bfr[n] = load_b_frag_t(&Bs[n * 16][0], 32);
#pragma unroll
    for (int n = 0; n < 4; ++n) acc[n] = wmma_bf16(a, bfr[n], acc[n]);
  }

  // ---- store C (ISA C-layout: lane l, vgpr r -> row r+8*(l>=16), col l&15) ----
  const int rbase = m0 + wv * 16 + ((lane >> 4) << 3);
  const int colb  = lane & 15;
#pragma unroll
  for (int n = 0; n < 4; ++n) {
#pragma unroll
    for (int r = 0; r < 8; ++r) {
      const int i = rbase + r;
      const int j = n0 + n * 16 + colb;
      const float v = acc[n][r];
      if (mode == 3) {
        Out[(size_t)i * EMB + j] = v;
      } else {
        const int bb = i >> 9, s = i & (SEQ - 1);
        const int h  = j >> 6, d = j & (DH - 1);
        if (mode == 0)
          Qh[(((size_t)bb * HEADS + h) * SEQ + s) * DH + d] = f2b(v);
        else if (mode == 1)
          Kh[(((size_t)bb * HEADS + h) * SEQ + s) * DH + d] = f2b(v);
        else
          VhT[(((size_t)bb * HEADS + h) * DH + d) * SEQ + s] = f2b(v);
      }
    }
  }
}

// ---------------------------------------------------------------------------
// qp[b,h,s,f] = sum_e qh[b,h,s,e] * p[b,s,h,e,f]
// One wave per (b,s,h). Streams the 64x64 f32 matrix (16KB) coalesced;
// memory-bound (268MB total) so prefetch ahead with global_prefetch_b8.
// ---------------------------------------------------------------------------
__global__ __launch_bounds__(256) void qp_kernel(
    const __bf16* __restrict__ Qh, const float* __restrict__ P,
    __bf16* __restrict__ QP) {
  __shared__ float qsh[8][64];
  const int wv   = threadIdx.x >> 5;
  const int lane = threadIdx.x & 31;
  const int idx  = blockIdx.x * 8 + wv;       // idx = (b*512 + s)*16 + h
  const int h = idx & 15;
  const int s = (idx >> 4) & (SEQ - 1);
  const int b = idx >> 13;

  const __bf16* qrow = Qh + (((size_t)b * HEADS + h) * SEQ + s) * DH;
  qsh[wv][lane]      = b2f(qrow[lane]);
  qsh[wv][lane + 32] = b2f(qrow[lane + 32]);
  __syncthreads();

  const float* p = P + (size_t)idx * (DH * DH);
  float acc0 = 0.f, acc1 = 0.f;
#pragma unroll
  for (int e = 0; e < DH; ++e) {
    if ((e & 15) == 0)
      __builtin_prefetch(p + e * DH + 1024, 0, 0);  // global_prefetch_b8
    const float qe = qsh[wv][e];
    acc0 = fmaf(qe, p[e * DH + lane], acc0);
    acc1 = fmaf(qe, p[e * DH + lane + 32], acc1);
  }

  __bf16* out = QP + (((size_t)b * HEADS + h) * SEQ + s) * DH;
  out[lane]      = f2b(acc0);
  out[lane + 32] = f2b(acc1);
}

// ---------------------------------------------------------------------------
// Flash-style causal attention per (b,h):
//   scores = (qp @ Kh^T) / 8 + causal;  att = softmax(scores) @ Vh
// Workgroup = 128 threads (4 waves); each wave owns a 16-row query tile; the
// waves share 32-key K/V chunks streamed into double-buffered LDS with
// global_load_async_to_lds_b128 (ASYNCcnt) so chunk i+1 loads overlap the
// WMMA/softmax work on chunk i. Online softmax keeps row stats in registers
// (C-layout rows live across 16-lane groups -> shfl_xor 1/2/4/8 reduce within
// a row). Probs round-trip LDS to convert C-layout -> A-layout for the V
// accumulation WMMAs.
// ---------------------------------------------------------------------------
__global__ __launch_bounds__(128) void attn(
    const __bf16* __restrict__ QP, const __bf16* __restrict__ Kh,
    const __bf16* __restrict__ VhT, __bf16* __restrict__ ATT) {
  __shared__ __bf16 Kc[2][32][64];   // keys   [t][d]  (double buffered)
  __shared__ __bf16 Vc[2][64][32];   // values [d][t]  (from VhT)
  __shared__ __bf16 Ps[4][16][32];

  const int tid  = threadIdx.x;
  const int wv   = tid >> 5;
  const int lane = tid & 31;
  const int bh   = blockIdx.y;          // b*16 + h
  const int b    = bh >> 4;
  const int h    = bh & 15;
  const int q0   = blockIdx.x * 64 + wv * 16;

  const __bf16* qpBase = QP + ((size_t)bh * SEQ + q0) * DH;
  const v16bf aq0 = load_a_frag(qpBase, DH);       // K = d 0..31
  const v16bf aq1 = load_a_frag(qpBase + 32, DH);  // K = d 32..63

  float mrow[8], lrow[8];
#pragma unroll
  for (int r = 0; r < 8; ++r) { mrow[r] = -1e30f; lrow[r] = 0.f; }
  v8f o[4];
#pragma unroll
  for (int n = 0; n < 4; ++n) o[n] = vzero8();

  const int colt  = lane & 15;
  const int rbase = (lane >> 4) << 3;
  const int tend  = blockIdx.x * 64 + 64;          // causal bound for this WG

  // K staging geometry: 32x64 bf16 = 4KB -> 2x b128 per thread
  const int krow = tid >> 2, kc0 = (tid & 3) * 16;
  // V staging geometry: 64x32 bf16 = 4KB -> 2x b128 per thread
  const int vrow = tid >> 1, vc0 = (tid & 1) * 16;

  // ---- prologue: async-stage chunk 0 into buffer 0 ----
  {
    const __bf16* ksrc = Kh + ((size_t)bh * SEQ + krow) * DH + kc0;
    async_copy16(&Kc[0][krow][kc0],     ksrc);
    async_copy16(&Kc[0][krow][kc0 + 8], ksrc + 8);
    const __bf16* vsrc = VhT + ((size_t)bh * DH + vrow) * SEQ + vc0;
    async_copy16(&Vc[0][vrow][vc0],     vsrc);
    async_copy16(&Vc[0][vrow][vc0 + 8], vsrc + 8);
  }

  int cur = 0;
  for (int t0 = 0; t0 < tend; t0 += 32, cur ^= 1) {
    wait_async();        // this wave's async copies for buffer `cur` done
    __syncthreads();     // ... and everyone else's too

    // ---- kick off async staging of the next chunk into the other buffer ----
    if (t0 + 32 < tend) {
      const int nxt = cur ^ 1;
      const __bf16* ksrc = Kh + ((size_t)bh * SEQ + t0 + 32 + krow) * DH + kc0;
      async_copy16(&Kc[nxt][krow][kc0],     ksrc);
      async_copy16(&Kc[nxt][krow][kc0 + 8], ksrc + 8);
      const __bf16* vsrc = VhT + ((size_t)bh * DH + vrow) * SEQ + t0 + 32 + vc0;
      async_copy16(&Vc[nxt][vrow][vc0],     vsrc);
      async_copy16(&Vc[nxt][vrow][vc0 + 8], vsrc + 8);
    }

    if (t0 <= q0 + 15) {   // wave participates (non-fully-masked chunk)
      // Preload all K fragments, then back-to-back score WMMAs.
      const v16bf kb00 = load_b_frag_t(&Kc[cur][0][0],   DH);
      const v16bf kb01 = load_b_frag_t(&Kc[cur][0][32],  DH);
      const v16bf kb10 = load_b_frag_t(&Kc[cur][16][0],  DH);
      const v16bf kb11 = load_b_frag_t(&Kc[cur][16][32], DH);
      v8f s0 = vzero8(), s1 = vzero8();
      s0 = wmma_bf16(aq0, kb00, s0);
      s0 = wmma_bf16(aq1, kb01, s0);
      s1 = wmma_bf16(aq0, kb10, s1);
      s1 = wmma_bf16(aq1, kb11, s1);

#pragma unroll
      for (int r = 0; r < 8; ++r) {
        const int srow = q0 + rbase + r;
        const float v0 = (t0 + colt      <= srow) ? s0[r] * 0.125f : -1e30f;
        const float v1 = (t0 + 16 + colt <= srow) ? s1[r] * 0.125f : -1e30f;

        float mx = fmaxf(v0, v1);                   // row-max across 16 lanes
        mx = fmaxf(mx, __shfl_xor(mx, 1));
        mx = fmaxf(mx, __shfl_xor(mx, 2));
        mx = fmaxf(mx, __shfl_xor(mx, 4));
        mx = fmaxf(mx, __shfl_xor(mx, 8));
        const float mnew  = fmaxf(mrow[r], mx);
        const float alpha = __expf(mrow[r] - mnew);
        const float p0 = __expf(v0 - mnew);
        const float p1 = __expf(v1 - mnew);
        float sum = p0 + p1;                        // row-sum across 16 lanes
        sum += __shfl_xor(sum, 1);
        sum += __shfl_xor(sum, 2);
        sum += __shfl_xor(sum, 4);
        sum += __shfl_xor(sum, 8);
        lrow[r] = lrow[r] * alpha + sum;
        mrow[r] = mnew;
#pragma unroll
        for (int n = 0; n < 4; ++n) o[n][r] *= alpha;

        Ps[wv][rbase + r][colt]      = f2b(p0);     // C-layout -> row-major
        Ps[wv][rbase + r][16 + colt] = f2b(p1);
      }

      const v16bf ap = load_a_frag(&Ps[wv][0][0], 32);   // A-layout over t
      v16bf vb[4];
#pragma unroll
      for (int n = 0; n < 4; ++n)
        vb[n] = load_b_frag_t(&Vc[cur][n * 16][0], 32);
#pragma unroll
      for (int n = 0; n < 4; ++n) o[n] = wmma_bf16(ap, vb[n], o[n]);
    }
  }

  // normalize and store att as bf16 matrix [b*S + s][h*64 + d]
#pragma unroll
  for (int n = 0; n < 4; ++n) {
#pragma unroll
    for (int r = 0; r < 8; ++r) {
      const int s = q0 + rbase + r;
      const int j = h * DH + n * 16 + colt;
      const float v = o[n][r] / lrow[r];
      ATT[((size_t)b * SEQ + s) * EMB + j] = f2b(v);
    }
  }
}

// ---------------------------------------------------------------------------
extern "C" void kernel_launch(void* const* d_in, const int* in_sizes, int n_in,
                              void* d_out, int out_size, void* d_ws,
                              size_t ws_size, hipStream_t stream) {
  (void)in_sizes; (void)n_in; (void)out_size; (void)ws_size;
  const float* q  = (const float*)d_in[0];
  const float* k  = (const float*)d_in[1];
  const float* v  = (const float*)d_in[2];
  const float* p  = (const float*)d_in[3];
  // d_in[4] = mask: causal mask is computed analytically in-kernel
  const float* Wq = (const float*)d_in[5];
  const float* Wk = (const float*)d_in[6];
  const float* Wv = (const float*)d_in[7];
  const float* Wo = (const float*)d_in[8];

  char* ws = (char*)d_ws;                 // 5 x 2MB bf16 intermediates
  __bf16* Qh  = (__bf16*)(ws + (0ull << 20));
  __bf16* Kh  = (__bf16*)(ws + (2ull << 20));
  __bf16* VhT = (__bf16*)(ws + (4ull << 20));
  __bf16* QPb = (__bf16*)(ws + (6ull << 20));
  __bf16* ATT = (__bf16*)(ws + (8ull << 20));
  float* out = (float*)d_out;

  // 1) Q/K/V projections (grid.z selects matrix)
  gemm_proj<<<dim3(8, 16, 3), 256, 0, stream>>>(
      q, k, v, ATT, Wq, Wk, Wv, Wo, Qh, Kh, VhT, out, 0);
  // 2) bilinear term qp = qh . p   (one wave per (b,s,h))
  qp_kernel<<<dim3(2048), 256, 0, stream>>>(Qh, p, QPb);
  // 3) causal flash attention -> ATT bf16
  attn<<<dim3(8, 32), 128, 0, stream>>>(QPb, Kh, VhT, ATT);
  // 4) output projection (f32 result)
  gemm_proj<<<dim3(8, 16, 1), 256, 0, stream>>>(
      q, k, v, ATT, Wq, Wk, Wv, Wo, Qh, Kh, VhT, out, 3);
}
